// Point_ContextualEncoder_6047313953110
// MI455X (gfx1250) — matrix operand order repaired and checked by reading, not compiled, not run on hardware
//
#include <hip/hip_runtime.h>

// ---------------------------------------------------------------------------
// CDNA5 (gfx1250) sparse InceptionResNet encoder.
// bf16 activations + v_wmma_f32_16x16x32_bf16, f32 scatter-add via
// global_atomic_add_f32. Memory-bound workload -> bf16 halves gather traffic.
// ---------------------------------------------------------------------------

typedef __attribute__((ext_vector_type(16))) __bf16 v16bf;
typedef __attribute__((ext_vector_type(8)))  __bf16 v8bf;
typedef __attribute__((ext_vector_type(8)))  float  v8f;

__device__ __forceinline__ unsigned short f2bf(float f) {
  unsigned u = __builtin_bit_cast(unsigned, f);
  u += 0x7FFFu + ((u >> 16) & 1u);            // round-to-nearest-even
  return (unsigned short)(u >> 16);
}

__device__ __forceinline__ v16bf cat8(v8bf lo, v8bf hi) {
  return __builtin_shufflevector(lo, hi, 0,1,2,3,4,5,6,7,8,9,10,11,12,13,14,15);
}

// ---------------------------------------------------------------------------
// Sparse conv: per (offset k, 16-row map tile) one wave does
//   gather A (16 x CINP bf16) -> WMMA -> atomic scatter-add f32.
// A layout (16-bit A, 16x32/Ktile): lane<16 : row=lane, K = {0..15}
//                                   lane>=16: row=lane-16, K = {16..31}
// B layout: lane group g, col n=lane&15 holds K0+16g .. K0+16g+15 of column n
//   -> weights stored transposed [k][n][cin], 32B contiguous per lane.
// D layout: vgpr v, group g -> row g*8+v, col = nt*16 + (lane&15).
// ---------------------------------------------------------------------------
template<int CINP, int COUTP, int COUT_REAL>
__global__ __launch_bounds__(128) void sconv_wmma(
    const unsigned short* __restrict__ F,     // [(n_rows+1) x CINP] bf16, zero pad row
    const unsigned short* __restrict__ Wt,    // [K x COUTP x CINP] bf16 (transposed)
    const int* __restrict__ in_map,           // [K x M]
    const int* __restrict__ out_map,          // [K x M]
    float* __restrict__ acc,                  // [n_out x COUT_REAL] f32
    int M, int in_pad, int out_pad)
{
  constexpr int KT = CINP / 32;
  constexpr int NT = COUTP / 16;
  const int lane = threadIdx.x & 31;
  const int g    = lane >> 4;
  const int ln   = lane & 15;
  const int wv   = threadIdx.x >> 5;
  const int m0   = (blockIdx.x * 4 + wv) * 16;
  if (m0 >= M) return;
  const int k = blockIdx.y;
  const int* im = in_map  + (size_t)k * M;
  const int* om = out_map + (size_t)k * M;

  const int mr   = m0 + ln;
  const int irow = (mr < M) ? im[mr] : in_pad;                 // pad -> zero row
  const unsigned short* arow  = F  + (size_t)irow * CINP;
  const unsigned short* wbase = Wt + (size_t)k * COUTP * CINP;

  v8f c[NT];
#pragma unroll
  for (int i = 0; i < NT; ++i) c[i] = {};

#pragma unroll
  for (int kt = 0; kt < KT; ++kt) {
    const int K0 = kt * 32;
    v8bf lo = *(const v8bf*)(arow + K0 + g * 16);
    v8bf hi = *(const v8bf*)(arow + K0 + g * 16 + 8);
    v16bf a = cat8(lo, hi);
#pragma unroll
    for (int nt = 0; nt < NT; ++nt) {
      const int n = nt * 16 + ln;
      v16bf b = *(const v16bf*)(wbase + (size_t)n * CINP + K0 + g * 16);
      c[nt] = __builtin_amdgcn_wmma_f32_16x16x32_bf16(
          false, a, false, b, (short)0, c[nt], false, false);
    }
  }

#pragma unroll
  for (int v = 0; v < 8; ++v) {
    const int mrr  = m0 + g * 8 + v;
    const int orow = (mrr < M) ? om[mrr] : out_pad;
    if (orow == out_pad) continue;                              // dropped row
#pragma unroll
    for (int nt = 0; nt < NT; ++nt) {
      const int col = nt * 16 + ln;
      if (col < COUT_REAL)
        atomicAdd(acc + (size_t)orow * COUT_REAL + col, c[nt][v]);
    }
  }
}

// ---------------------------------------------------------------------------
// Dense 1x1 conv (identity map, contiguous rows) with fused bias/ReLU/convert.
// ---------------------------------------------------------------------------
template<int CINP, int COUTP, bool RELU, bool OUT_BF16, int OUT_STRIDE, bool PAD_HI>
__global__ __launch_bounds__(128) void dense_wmma(
    const unsigned short* __restrict__ F,     // [(nrows+1) x CINP] bf16
    const unsigned short* __restrict__ Wt,    // [COUTP x CINP] bf16 (transposed)
    const float* __restrict__ bias,
    void* __restrict__ outp, int nrows)
{
  constexpr int KT = CINP / 32;
  constexpr int NT = COUTP / 16;
  const int lane = threadIdx.x & 31;
  const int g    = lane >> 4;
  const int ln   = lane & 15;
  const int wv   = threadIdx.x >> 5;
  const int m0   = (blockIdx.x * 4 + wv) * 16;
  if (m0 >= nrows) return;

  const int mr   = m0 + ln;
  const int irow = (mr < nrows) ? mr : nrows;                   // zero pad row
  const unsigned short* arow = F + (size_t)irow * CINP;

  v8f c[NT];
#pragma unroll
  for (int i = 0; i < NT; ++i) c[i] = {};

#pragma unroll
  for (int kt = 0; kt < KT; ++kt) {
    const int K0 = kt * 32;
    v8bf lo = *(const v8bf*)(arow + K0 + g * 16);
    v8bf hi = *(const v8bf*)(arow + K0 + g * 16 + 8);
    v16bf a = cat8(lo, hi);
#pragma unroll
    for (int nt = 0; nt < NT; ++nt) {
      const int n = nt * 16 + ln;
      v16bf b = *(const v16bf*)(Wt + (size_t)n * CINP + K0 + g * 16);
      c[nt] = __builtin_amdgcn_wmma_f32_16x16x32_bf16(
          false, a, false, b, (short)0, c[nt], false, false);
    }
  }

#pragma unroll
  for (int v = 0; v < 8; ++v) {
    const int row = m0 + g * 8 + v;
    if (row >= nrows) continue;
#pragma unroll
    for (int nt = 0; nt < NT; ++nt) {
      const int col = nt * 16 + ln;
      float t = c[nt][v] + bias[col];
      if (RELU) t = t > 0.f ? t : 0.f;
      if (OUT_BF16) {
        unsigned short* o = (unsigned short*)outp;
        o[(size_t)row * OUT_STRIDE + col] = f2bf(t);
        if (PAD_HI) o[(size_t)row * OUT_STRIDE + COUTP + col] = 0;  // zero K-pad cols
      } else {
        float* o = (float*)outp;
        o[(size_t)row * OUT_STRIDE + col] = t;
      }
    }
  }
}

// ----------------------------- elementwise ---------------------------------

// W[k][cin][cout] f32 -> Wt[k][n][c] bf16 with zero padding.
__global__ void k_conv_w(const float* __restrict__ src, unsigned short* __restrict__ dst,
                         int K, int CIN, int COUT, int CINP, int COUTP) {
  size_t tot = (size_t)K * COUTP * CINP;
  size_t i = (size_t)blockIdx.x * blockDim.x + threadIdx.x;
  if (i >= tot) return;
  int cc = (int)(i % CINP);
  size_t r = i / CINP;
  int n = (int)(r % COUTP);
  int k = (int)(r / COUTP);
  float v = 0.f;
  if (cc < CIN && n < COUT) v = src[((size_t)k * CIN + cc) * COUT + n];
  dst[i] = f2bf(v);
}

// fbf[r][0:64]=x_F, [64:128]=ctx_F, row NI = zeros.
__global__ void k_prep(const float* __restrict__ xF, const float* __restrict__ ctxF,
                       unsigned short* __restrict__ fbf, int NI) {
  size_t tot = (size_t)(NI + 1) * 128;
  size_t i = (size_t)blockIdx.x * blockDim.x + threadIdx.x;
  if (i >= tot) return;
  int cc = (int)(i & 127);
  size_t r = i >> 7;
  float v = 0.f;
  if ((long)r < NI) v = (cc < 64) ? xF[r * 64 + cc] : ctxF[r * 64 + cc - 64];
  fbf[i] = f2bf(v);
}

// h += b1 (in place, keeps f32 for residual) and write bf16 copy (+zero pad row).
__global__ void k_epi1(float* __restrict__ hacc, const float* __restrict__ b1,
                       unsigned short* __restrict__ hbf, int NO) {
  size_t tot = (size_t)(NO + 1) * 64;
  size_t i = (size_t)blockIdx.x * blockDim.x + threadIdx.x;
  if (i >= tot) return;
  int cc = (int)(i & 63);
  size_t r = i >> 6;
  if ((long)r < NO) {
    float v = hacc[i] + b1[cc];
    hacc[i] = v;
    hbf[i] = f2bf(v);
  } else hbf[i] = 0;
}

// bf16(relu(acc16 + bias)) into stride-32 buffer, hi half + pad row zero.
__global__ void k_epi_relu16(const float* __restrict__ acc, const float* __restrict__ bias,
                             unsigned short* __restrict__ out, int NO) {
  size_t tot = (size_t)(NO + 1) * 32;
  size_t i = (size_t)blockIdx.x * blockDim.x + threadIdx.x;
  if (i >= tot) return;
  int cc = (int)(i & 31);
  size_t r = i >> 5;
  unsigned short v = 0;
  if ((long)r < NO && cc < 16) {
    float t = acc[r * 16 + cc] + bias[cc];
    v = f2bf(t > 0.f ? t : 0.f);
  }
  out[i] = v;
}

// h2 = cat(out0+br01, path1) + h ; write bf16 (reuses h_bf storage).
__global__ void k_epi_merge(const float* __restrict__ out0acc, const float* __restrict__ br01,
                            const float* __restrict__ path1, const float* __restrict__ h,
                            unsigned short* __restrict__ h2bf, int NO) {
  size_t tot = (size_t)(NO + 1) * 64;
  size_t i = (size_t)blockIdx.x * blockDim.x + threadIdx.x;
  if (i >= tot) return;
  int cc = (int)(i & 63);
  size_t r = i >> 6;
  unsigned short v = 0;
  if ((long)r < NO) {
    float t = (cc < 32) ? (out0acc[r * 32 + cc] + br01[cc]) : path1[r * 32 + (cc - 32)];
    v = f2bf(t + h[i]);
  }
  h2bf[i] = v;
}

__global__ void k_epi_out(const float* __restrict__ acc, const float* __restrict__ b2,
                          float* __restrict__ out, int NO) {
  size_t tot = (size_t)NO * 8;
  size_t i = (size_t)blockIdx.x * blockDim.x + threadIdx.x;
  if (i >= tot) return;
  out[i] = acc[i] + b2[i & 7];
}

// ---------------------------------------------------------------------------

static inline char* carve(char*& p, size_t bytes) {
  char* r = p;
  p += (bytes + 255) & ~(size_t)255;
  return r;
}
static inline long cdivl(long a, long b) { return (a + b - 1) / b; }

extern "C" void kernel_launch(void* const* d_in, const int* in_sizes, int n_in,
                              void* d_out, int out_size, void* d_ws, size_t ws_size,
                              hipStream_t stream) {
  const float* xF   = (const float*)d_in[0];
  const float* ctxF = (const float*)d_in[1];
  const float* W1   = (const float*)d_in[2];
  const float* b1   = (const float*)d_in[3];
  const float* Wr00 = (const float*)d_in[4];
  const float* br00 = (const float*)d_in[5];
  const float* Wr01 = (const float*)d_in[6];
  const float* br01 = (const float*)d_in[7];
  const float* Wr10 = (const float*)d_in[8];
  const float* br10 = (const float*)d_in[9];
  const float* Wr11 = (const float*)d_in[10];
  const float* br11 = (const float*)d_in[11];
  const float* Wr12 = (const float*)d_in[12];
  const float* br12 = (const float*)d_in[13];
  const float* W2   = (const float*)d_in[14];
  const float* b2   = (const float*)d_in[15];
  const int* in1    = (const int*)d_in[16];
  const int* out1   = (const int*)d_in[17];
  const int* in3    = (const int*)d_in[18];
  const int* out3   = (const int*)d_in[19];
  (void)n_in; (void)ws_size;

  const int NI = in_sizes[0] / 64;          // n_in points
  const int M1 = in_sizes[16] / 8;          // k=2 map width
  const int M3 = in_sizes[18] / 27;         // k=3 map width
  const int NO = out_size / 8;              // n_out (out is [n_out, 8] f32)

  // ---- workspace carve (bf16 activations, f32 accumulators) ----
  char* p = (char*)d_ws;
  unsigned short* fbf   = (unsigned short*)carve(p, (size_t)(NI + 1) * 128 * 2);
  unsigned short* W1t   = (unsigned short*)carve(p, (size_t)8 * 64 * 128 * 2);
  unsigned short* Wr00t = (unsigned short*)carve(p, (size_t)27 * 16 * 64 * 2);
  unsigned short* Wr01t = (unsigned short*)carve(p, (size_t)27 * 32 * 32 * 2);
  unsigned short* Wr10t = (unsigned short*)carve(p, (size_t)16 * 64 * 2);
  unsigned short* Wr11t = (unsigned short*)carve(p, (size_t)27 * 16 * 32 * 2);
  unsigned short* Wr12t = (unsigned short*)carve(p, (size_t)32 * 32 * 2);
  unsigned short* W2t   = (unsigned short*)carve(p, (size_t)27 * 16 * 64 * 2);
  float*          hacc  = (float*)carve(p, (size_t)NO * 64 * 4);        // -> h (f32)
  unsigned short* hbf   = (unsigned short*)carve(p, (size_t)(NO + 1) * 64 * 2);
  float*          acc16 = (float*)carve(p, (size_t)NO * 16 * 4);
  unsigned short* t0bf  = (unsigned short*)carve(p, (size_t)(NO + 1) * 32 * 2);
  float*          acc32 = (float*)carve(p, (size_t)NO * 32 * 4);        // out0 acc
  unsigned short* tAbf  = (unsigned short*)carve(p, (size_t)(NO + 1) * 32 * 2);
  unsigned short* t2bf  = (unsigned short*)carve(p, (size_t)(NO + 1) * 32 * 2);
  float*          path1 = (float*)carve(p, (size_t)NO * 32 * 4);
  float*          oacc  = (float*)carve(p, (size_t)NO * 8 * 4);

  const int T = 256;
  auto ge = [&](size_t tot) { return dim3((unsigned)cdivl((long)tot, T)); };

  // ---- stage 0: prep activations + weights ----
  k_prep<<<ge((size_t)(NI + 1) * 128), T, 0, stream>>>(xF, ctxF, fbf, NI);
  k_conv_w<<<ge((size_t)8 * 64 * 128), T, 0, stream>>>(W1,   W1t,   8, 128, 64, 128, 64);
  k_conv_w<<<ge((size_t)27 * 16 * 64), T, 0, stream>>>(Wr00, Wr00t, 27, 64, 16, 64, 16);
  k_conv_w<<<ge((size_t)27 * 32 * 32), T, 0, stream>>>(Wr01, Wr01t, 27, 16, 32, 32, 32);
  k_conv_w<<<ge((size_t)16 * 64),      T, 0, stream>>>(Wr10, Wr10t, 1, 64, 16, 64, 16);
  k_conv_w<<<ge((size_t)27 * 16 * 32), T, 0, stream>>>(Wr11, Wr11t, 27, 16, 16, 32, 16);
  k_conv_w<<<ge((size_t)32 * 32),      T, 0, stream>>>(Wr12, Wr12t, 1, 16, 32, 32, 32);
  k_conv_w<<<ge((size_t)27 * 16 * 64), T, 0, stream>>>(W2,   W2t,   27, 64, 8, 64, 16);

  const int t1 = (int)cdivl(M1, 16), t3 = (int)cdivl(M3, 16), td = (int)cdivl(NO, 16);

  // ---- conv1: 128 -> 64, 8 offsets ----
  hipMemsetAsync(hacc, 0, (size_t)NO * 64 * 4, stream);
  sconv_wmma<128, 64, 64><<<dim3((unsigned)cdivl(t1, 4), 8), 128, 0, stream>>>(
      fbf, W1t, in1, out1, hacc, M1, NI, NO);
  k_epi1<<<ge((size_t)(NO + 1) * 64), T, 0, stream>>>(hacc, b1, hbf, NO);

  // ---- inception path0: sconv 64->16 relu, sconv 16->32 ----
  hipMemsetAsync(acc16, 0, (size_t)NO * 16 * 4, stream);
  sconv_wmma<64, 16, 16><<<dim3((unsigned)cdivl(t3, 4), 27), 128, 0, stream>>>(
      hbf, Wr00t, in3, out3, acc16, M3, NO, NO);
  k_epi_relu16<<<ge((size_t)(NO + 1) * 32), T, 0, stream>>>(acc16, br00, t0bf, NO);
  hipMemsetAsync(acc32, 0, (size_t)NO * 32 * 4, stream);
  sconv_wmma<32, 32, 32><<<dim3((unsigned)cdivl(t3, 4), 27), 128, 0, stream>>>(
      t0bf, Wr01t, in3, out3, acc32, M3, NO, NO);

  // ---- inception path1: 1x1 relu, sconv 16->16 relu, 1x1 ----
  hipMemsetAsync(tAbf + (size_t)NO * 32, 0, 64, stream);   // zero pad row
  dense_wmma<64, 16, true, true, 32, true><<<dim3((unsigned)cdivl(td, 4)), 128, 0, stream>>>(
      hbf, Wr10t, br10, tAbf, NO);
  hipMemsetAsync(acc16, 0, (size_t)NO * 16 * 4, stream);
  sconv_wmma<32, 16, 16><<<dim3((unsigned)cdivl(t3, 4), 27), 128, 0, stream>>>(
      tAbf, Wr11t, in3, out3, acc16, M3, NO, NO);
  k_epi_relu16<<<ge((size_t)(NO + 1) * 32), T, 0, stream>>>(acc16, br11, t2bf, NO);
  dense_wmma<32, 32, false, false, 32, false><<<dim3((unsigned)cdivl(td, 4)), 128, 0, stream>>>(
      t2bf, Wr12t, br12, path1, NO);

  // ---- merge: h2 = cat(out0, path1) + h  (reuse hbf storage) ----
  k_epi_merge<<<ge((size_t)(NO + 1) * 64), T, 0, stream>>>(acc32, br01, path1, hacc, hbf, NO);

  // ---- conv2: 64 -> 8 (N-padded to 16) ----
  hipMemsetAsync(oacc, 0, (size_t)NO * 8 * 4, stream);
  sconv_wmma<64, 16, 8><<<dim3((unsigned)cdivl(t3, 4), 27), 128, 0, stream>>>(
      hbf, W2t, in3, out3, oacc, M3, NO, NO);
  k_epi_out<<<ge((size_t)NO * 8), T, 0, stream>>>(oacc, b2, (float*)d_out, NO);
}
